// CGLSTMCellv1_42683384987798
// MI455X (gfx1250) — matrix operand order repaired and checked by reading, not compiled, not run on hardware
//
#include <hip/hip_runtime.h>
#include <hip/hip_bf16.h>
#include <cstdint>
#include <cstddef>

typedef __attribute__((ext_vector_type(16))) __bf16 v16bf;
typedef __attribute__((ext_vector_type(8)))  __bf16 v8bf;
typedef __attribute__((ext_vector_type(8)))  float  v8f;

#define COS_EPS 1e-6f
#define LN_EPS  1e-5f

// ---------------------------------------------------------------------------
// Kernel 1: pack x and hx (f32) into bf16 concat buffer [B, 2048]
// ---------------------------------------------------------------------------
__global__ __launch_bounds__(256)
void k_pack_xh(const float* __restrict__ x, const float* __restrict__ hx,
               __bf16* __restrict__ concat, int n) {
    int idx = blockIdx.x * 256 + threadIdx.x;
    if (idx >= n) return;
    int b = idx >> 10, i = idx & 1023;
    size_t base = (size_t)b * 2048;
    concat[base + i]        = (__bf16)x[idx];
    concat[base + 1024 + i] = (__bf16)hx[idx];
}

// ---------------------------------------------------------------------------
// Kernel 2: tiled transpose f32[R,C] -> bf16[C,R]
// ---------------------------------------------------------------------------
__global__ __launch_bounds__(256)
void k_transpose_cvt(const float* __restrict__ src, __bf16* __restrict__ dst,
                     int R, int C) {
    __shared__ float tile[32][33];
    int c0 = blockIdx.x * 32, r0 = blockIdx.y * 32;
    int tx = threadIdx.x, ty = threadIdx.y;   // 32 x 8
#pragma unroll
    for (int i = 0; i < 32; i += 8)
        tile[ty + i][tx] = src[(size_t)(r0 + ty + i) * C + (c0 + tx)];
    __syncthreads();
#pragma unroll
    for (int i = 0; i < 32; i += 8)
        dst[(size_t)(c0 + ty + i) * R + (r0 + tx)] = (__bf16)tile[tx][ty + i];
}

// ---------------------------------------------------------------------------
// Kernel 3/5: bf16 WMMA GEMM.  D[M,N](f32) = A[M,K](bf16, lda) * Bt[N,K]^T + bias
// Block = 128 threads (4 waves, 2x2), block tile 128(M) x 128(N),
// wave tile 64x64 -> 4x4 accumulators, K-step 32:
//   16 v_wmma per 16 x global_load_b128 (1.0 WMMA/load), 4x fragment reuse.
// Fragment layouts per CDNA5 ISA 7.12.2 (wave32):
//   A 16x32: lanes 0-15 row M=r hold K {k0..k0+7, k0+16..k0+23},
//            lanes 16-31 hold K {k0+8..k0+15, k0+24..k0+31}
//   B 32x16: lane col N=r; lanes 0-15 hold K {k0..k0+7, k0+16..k0+23}, lanes 16-31 {k0+8..k0+15, k0+24..k0+31}
//   C/D 16x16 f32: elem e -> row = e + 8*(lane/16), col = lane%16
// ---------------------------------------------------------------------------
__global__ __launch_bounds__(128)
void k_gemm_bf16(const __bf16* __restrict__ A, const __bf16* __restrict__ Bt,
                 const float* __restrict__ bias, float* __restrict__ D,
                 int K, int lda, int ldb, int ldd) {
    const int lane = threadIdx.x & 31;
    const int wave = threadIdx.x >> 5;
    const int half = lane >> 4;
    const int r    = lane & 15;
    const int mbase = blockIdx.y * 128 + (wave >> 1) * 64;
    const int nbase = blockIdx.x * 128 + (wave & 1)  * 64;

    const __bf16* pa0 = A  + (size_t)(mbase + r) * lda + half * 8;
    const __bf16* pb0 = Bt + (size_t)(nbase + r) * ldb + half * 8;
    const size_t astep = (size_t)16 * lda;   // next 16-row A tile
    const size_t bstep = (size_t)16 * ldb;   // next 16-col B tile

    v8f acc[4][4] = {};

    for (int k0 = 0; k0 < K; k0 += 32) {
        v16bf afrag[4];
#pragma unroll
        for (int i = 0; i < 4; ++i) {
            const __bf16* pa = pa0 + (size_t)i * astep + k0;
            v8bf lo = *(const v8bf*)pa;          // K = base .. base+7
            v8bf hi = *(const v8bf*)(pa + 16);   // K = base+16 .. base+23
            afrag[i] = __builtin_shufflevector(lo, hi,
                0, 1, 2, 3, 4, 5, 6, 7, 8, 9, 10, 11, 12, 13, 14, 15);
        }
        v16bf bfrag[4];
#pragma unroll
        for (int j = 0; j < 4; ++j) {
            const __bf16* pb = pb0 + (size_t)j * bstep + k0;
            v8bf lo = *(const v8bf*)pb;
            v8bf hi = *(const v8bf*)(pb + 16);
            bfrag[j] = __builtin_shufflevector(lo, hi,
                0, 1, 2, 3, 4, 5, 6, 7, 8, 9, 10, 11, 12, 13, 14, 15);
        }
#pragma unroll
        for (int i = 0; i < 4; ++i)
#pragma unroll
            for (int j = 0; j < 4; ++j)
                acc[i][j] = __builtin_amdgcn_wmma_f32_16x16x32_bf16(
                    false, afrag[i], false, bfrag[j],
                    (short)0, acc[i][j], false, false);
    }

#pragma unroll
    for (int i = 0; i < 4; ++i) {
        const int rowbase = mbase + i * 16 + half * 8;
#pragma unroll
        for (int j = 0; j < 4; ++j) {
            const int col = nbase + j * 16 + r;
            const float bv = bias[col];
#pragma unroll
            for (int e = 0; e < 8; ++e)
                D[(size_t)(rowbase + e) * ldd + col] = acc[i][j][e] + bv;
        }
    }
}

// ---------------------------------------------------------------------------
// Kernel 4a: attn[b] = sigmoid(cos_sim(input_mapped[b], hx[b]))
// ---------------------------------------------------------------------------
__global__ __launch_bounds__(256)
void k_attn(const float* __restrict__ im, const float* __restrict__ hx,
            float* __restrict__ attn) {
    __shared__ float s0[256], s1[256], s2[256];
    const int b = blockIdx.x, t = threadIdx.x;
    const float* a = im + (size_t)b * 1024;
    const float* h = hx + (size_t)b * 1024;
    float dot = 0.f, na = 0.f, nb = 0.f;
#pragma unroll
    for (int e = 0; e < 4; ++e) {
        float av = a[t * 4 + e], hv = h[t * 4 + e];
        dot += av * hv; na += av * av; nb += hv * hv;
    }
    s0[t] = dot; s1[t] = na; s2[t] = nb; __syncthreads();
    for (int off = 128; off > 0; off >>= 1) {
        if (t < off) { s0[t] += s0[t + off]; s1[t] += s1[t + off]; s2[t] += s2[t + off]; }
        __syncthreads();
    }
    if (t == 0) {
        float c = s0[0] / (fmaxf(sqrtf(s1[0]), COS_EPS) * fmaxf(sqrtf(s2[0]), COS_EPS));
        attn[b] = 1.f / (1.f + expf(-c));
    }
}

// ---------------------------------------------------------------------------
// Kernel 4b: concat[:, :1024] = bf16(x * (1 + attn))
// ---------------------------------------------------------------------------
__global__ __launch_bounds__(256)
void k_modx(const float* __restrict__ x, const float* __restrict__ attn,
            __bf16* __restrict__ concat, int n) {
    int idx = blockIdx.x * 256 + threadIdx.x;
    if (idx >= n) return;
    int b = idx >> 10, i = idx & 1023;
    concat[(size_t)b * 2048 + i] = (__bf16)(x[idx] * (1.f + attn[b]));
}

// ---------------------------------------------------------------------------
// Kernel 6: per-row epilogue — 4 layernorms + activations + cell update +
// cosine output gate.  One 256-thread block per row, 4 elems/thread/gate.
// ---------------------------------------------------------------------------
__global__ __launch_bounds__(256)
void k_pointwise(const float* __restrict__ gates, const float* __restrict__ cx,
                 const float* __restrict__ gammas, const float* __restrict__ betas,
                 float* __restrict__ out_h, float* __restrict__ out_c) {
    __shared__ float s0[256], s1[256], s2[256];
    const int b = blockIdx.x, t = threadIdx.x;
    const float* grow = gates + (size_t)b * 4096;

    float act[4][4];
#pragma unroll
    for (int g = 0; g < 4; ++g) {
        float v[4], lsum = 0.f, lsq = 0.f;
#pragma unroll
        for (int e = 0; e < 4; ++e) {
            v[e] = grow[g * 1024 + t * 4 + e];
            lsum += v[e]; lsq += v[e] * v[e];
        }
        s0[t] = lsum; s1[t] = lsq; __syncthreads();
        for (int off = 128; off > 0; off >>= 1) {
            if (t < off) { s0[t] += s0[t + off]; s1[t] += s1[t + off]; }
            __syncthreads();
        }
        const float mu  = s0[0] * (1.f / 1024.f);
        const float var = s1[0] * (1.f / 1024.f) - mu * mu;
        __syncthreads();
        const float rstd = rsqrtf(var + LN_EPS);
#pragma unroll
        for (int e = 0; e < 4; ++e) {
            const int j = t * 4 + e;
            const float y = (v[e] - mu) * rstd * gammas[g * 1024 + j] + betas[g * 1024 + j];
            act[g][e] = (g == 2) ? tanhf(y) : 1.f / (1.f + expf(-y));
        }
    }

    float cn[4], hn[4];
    float dot = 0.f, nh = 0.f, nc = 0.f;
#pragma unroll
    for (int e = 0; e < 4; ++e) {
        const int j = t * 4 + e;
        const float co = cx[(size_t)b * 1024 + j];
        cn[e] = act[1][e] * co + act[0][e] * act[2][e];
        hn[e] = act[3][e] * tanhf(cn[e]);
        dot += hn[e] * cn[e]; nh += hn[e] * hn[e]; nc += cn[e] * cn[e];
    }
    s0[t] = dot; s1[t] = nh; s2[t] = nc; __syncthreads();
    for (int off = 128; off > 0; off >>= 1) {
        if (t < off) { s0[t] += s0[t + off]; s1[t] += s1[t + off]; s2[t] += s2[t + off]; }
        __syncthreads();
    }
    const float cosv = s0[0] / (fmaxf(sqrtf(s1[0]), COS_EPS) * fmaxf(sqrtf(s2[0]), COS_EPS));
    const float gco  = 1.f / (1.f + expf(-(cosv + 1.f) * 0.5f));
#pragma unroll
    for (int e = 0; e < 4; ++e) {
        const int j = t * 4 + e;
        out_h[(size_t)b * 1024 + j] = hn[e] * (1.f + gco);
        out_c[(size_t)b * 1024 + j] = cn[e];
    }
}

// ---------------------------------------------------------------------------
extern "C" void kernel_launch(void* const* d_in, const int* in_sizes, int n_in,
                              void* d_out, int out_size, void* d_ws, size_t ws_size,
                              hipStream_t stream) {
    const float* x  = (const float*)d_in[0];
    const float* hx = (const float*)d_in[1];
    const float* cx = (const float*)d_in[2];
    const float* W  = (const float*)d_in[3];
    const float* bG = (const float*)d_in[4];
    const float* Wm = (const float*)d_in[5];
    const float* bm = (const float*)d_in[6];
    const float* gammas = (const float*)d_in[7];
    const float* betas  = (const float*)d_in[8];

    const int I = 1024, H = 1024;
    const int B = in_sizes[0] / I;      // 8192

    char* ws = (char*)d_ws;
    size_t off = 0;
    auto carve = [&](size_t bytes) -> char* {
        char* p = ws + off;
        off = (off + bytes + 255) & ~(size_t)255;
        return p;
    };
    __bf16* concat = (__bf16*)carve((size_t)B * 2048 * 2);   // [B, 2048] bf16
    __bf16* Wt     = (__bf16*)carve((size_t)4096 * 2048 * 2);// W transposed, bf16
    __bf16* Wmt    = (__bf16*)carve((size_t)1024 * 1024 * 2);// Wm transposed, bf16
    float*  im     = (float*)carve((size_t)B * 1024 * 4);    // input_mapped
    float*  attn   = (float*)carve((size_t)B * 4);
    float*  gbuf   = (float*)carve((size_t)B * 4096 * 4);    // gates

    float* out_h = (float*)d_out;
    float* out_c = out_h + (size_t)B * H;

    // 1) pack x, hx -> bf16 concat
    k_pack_xh<<<(B * I + 255) / 256, 256, 0, stream>>>(x, hx, concat, B * I);

    // 2) transpose-convert weights (L2-resident afterwards)
    k_transpose_cvt<<<dim3(4096 / 32, 2048 / 32), dim3(32, 8), 0, stream>>>(W,  Wt,  2048, 4096);
    k_transpose_cvt<<<dim3(1024 / 32, 1024 / 32), dim3(32, 8), 0, stream>>>(Wm, Wmt, 1024, 1024);

    // 3) GEMM1: im = x @ Wm + bm   (A = concat first 1024 cols, lda = 2048)
    k_gemm_bf16<<<dim3(1024 / 128, B / 128), 128, 0, stream>>>(
        concat, Wmt, bm, im, /*K=*/1024, /*lda=*/2048, /*ldb=*/1024, /*ldd=*/1024);

    // 4) attention gate + x_mod rewrite
    k_attn<<<B, 256, 0, stream>>>(im, hx, attn);
    k_modx<<<(B * I + 255) / 256, 256, 0, stream>>>(x, attn, concat, B * I);

    // 5) GEMM2: gates = [x_mod, hx] @ W + b
    k_gemm_bf16<<<dim3(4096 / 128, B / 128), 128, 0, stream>>>(
        concat, Wt, bG, gbuf, /*K=*/2048, /*lda=*/2048, /*ldb=*/2048, /*ldd=*/4096);

    // 6) layernorms + activations + cell update + cosine output gate
    k_pointwise<<<B, 256, 0, stream>>>(gbuf, cx, gammas, betas, out_h, out_c);

    (void)n_in; (void)out_size; (void)ws_size;
}
